// DeformSpaceAttentionv2_26079041421809
// MI455X (gfx1250) — matrix operand order repaired and checked by reading, not compiled, not run on hardware
//
#include <hip/hip_runtime.h>
#include <hip/hip_bf16.h>

#define BB   2
#define CCH  256
#define HH   128
#define WWD  128
#define HWSZ (HH * WWD)
#define C4O  64
#define BN_EPS_F 1e-5f

typedef __attribute__((ext_vector_type(16))) _Float16     v16h;
typedef __attribute__((ext_vector_type(8)))  float        v8f;
typedef __attribute__((ext_vector_type(4)))  float        f32x4;
typedef __attribute__((ext_vector_type(4)))  unsigned int u32x4;

union ABu { u32x4 q[2]; v16h h; };
union H2  { _Float16 h[2]; unsigned int u; };

__launch_bounds__(128, 1)
__global__ void deform_attn_wmma_kernel(
    const float* __restrict__ x,
    const float* __restrict__ offset,
    const float* __restrict__ W0,
    const float* __restrict__ b0,
    const float* __restrict__ gamma,
    const float* __restrict__ beta,
    const float* __restrict__ run_mean,
    const float* __restrict__ run_var,
    const float* __restrict__ W1,
    const float* __restrict__ b1,
    float* __restrict__ out)
{
    // feat tile: [wave][pos 0..15][channel-pair 0..127] as packed f16x2 -> 32 KB
    __shared__ unsigned int feat_lds[4 * 16 * 128];
    // attention: [wave][pos 0..15][group 0..3]
    __shared__ float att_lds[4 * 16 * 4];

    const int tid  = threadIdx.x;
    const int wv   = tid >> 5;        // wave in block (0..3)
    const int lane = tid & 31;
    const int ln   = lane & 15;       // position-within-wave / matrix column index
    const int half = lane >> 4;       // channel-half / K-half selector

    const int bid = blockIdx.x;       // 0 .. B*H*2-1
    const int b   = bid / (HH * 2);
    const int rem = bid - b * (HH * 2);
    const int y   = rem >> 1;
    const int x0  = (rem & 1) * 64 + wv * 16;
    const int xg  = x0 + ln;          // this lane's spatial x

    // ---------------- Phase 1: deformable 3x3 bilinear gather + max over 9 taps --
    int   goff[36];
    float gw[36];
    {
        const int obase = (b * 18) * HWSZ + y * WWD + xg;
        #pragma unroll
        for (int k = 0; k < 9; ++k) {
            const float oy = offset[obase + (2 * k) * HWSZ];
            const float ox = offset[obase + (2 * k + 1) * HWSZ];
            const float py  = (float)y  + (float)(k / 3 - 1) + oy;
            const float pxf = (float)xg + (float)(k % 3 - 1) + ox;
            const float fy0 = floorf(py);
            const float fx0 = floorf(pxf);
            const float ay = py - fy0;
            const float ax = pxf - fx0;
            #pragma unroll
            for (int dy = 0; dy < 2; ++dy) {
                #pragma unroll
                for (int dx = 0; dx < 2; ++dx) {
                    const float yi = fy0 + (float)dy;
                    const float xi = fx0 + (float)dx;
                    const float wgt = (dy ? ay : 1.f - ay) * (dx ? ax : 1.f - ax);
                    const bool valid = (yi >= 0.f) && (yi <= (float)(HH - 1)) &&
                                       (xi >= 0.f) && (xi <= (float)(WWD - 1));
                    const int yc = (int)fminf(fmaxf(yi, 0.f), (float)(HH - 1));
                    const int xc = (int)fminf(fmaxf(xi, 0.f), (float)(WWD - 1));
                    goff[k * 4 + dy * 2 + dx] = yc * WWD + xc;
                    gw  [k * 4 + dy * 2 + dx] = valid ? wgt : 0.f;
                }
            }
        }
    }

    {
        // lane covers 128 contiguous channels (half selects which 128)
        const float* xb = x + (size_t)b * CCH * HWSZ + (size_t)(half * 128) * HWSZ;
        const int fbase = (wv * 16 + ln) * 128 + half * 64;   // channel-pair units
        for (int i = 0; i < 64; ++i) {
            const float* p0 = xb + (size_t)(2 * i) * HWSZ;
            const float* p1 = p0 + HWSZ;
            float m0 = -3.0e38f, m1 = -3.0e38f;
            #pragma unroll
            for (int k = 0; k < 9; ++k) {
                float v0 = 0.f, v1 = 0.f;
                #pragma unroll
                for (int t = 0; t < 4; ++t) {
                    const int   o  = goff[k * 4 + t];
                    const float wq = gw[k * 4 + t];
                    v0 = fmaf(wq, p0[o], v0);
                    v1 = fmaf(wq, p1[o], v1);
                }
                m0 = fmaxf(m0, v0);
                m1 = fmaxf(m1, v1);
            }
            H2 pk;
            pk.h[0] = (_Float16)m0;
            pk.h[1] = (_Float16)m1;
            feat_lds[fbase + i] = pk.u;
        }
    }
    __syncthreads();

    // ---------------- Phase 2: (16x256)x(256x64) per wave via v_wmma_f32_16x16x32_f16
    v8f acc[4];
    #pragma unroll
    for (int nb = 0; nb < 4; ++nb) {
        #pragma unroll
        for (int r = 0; r < 8; ++r) acc[nb][r] = 0.f;
    }

    #pragma unroll
    for (int kc = 0; kc < 8; ++kc) {       // K = kc*32 .. kc*32+31
        // A-matrix (contiguous k order, same as B): lane row M=ln,
        // VGPR0-7: K = kc32 + half*16 + {0..15}
        ABu a;
        const u32x4* ap = (const u32x4*)&feat_lds[(wv * 16 + ln) * 128 + kc * 16 + half * 8];
        a.q[0] = ap[0];
        a.q[1] = ap[1];

        #pragma unroll
        for (int nb = 0; nb < 4; ++nb) {
            // B-matrix 32x16: lane col N=ln -> o = nb*16+ln; lanes 0-15 K 0-15, 16-31 K 16-31
            const f32x4* wp = (const f32x4*)(W0 + (size_t)(nb * 16 + ln) * 256
                                             + kc * 32 + half * 16);
            v16h bh;
            #pragma unroll
            for (int q = 0; q < 4; ++q) {
                const f32x4 f = wp[q];
                bh[4 * q + 0] = (_Float16)f[0];
                bh[4 * q + 1] = (_Float16)f[1];
                bh[4 * q + 2] = (_Float16)f[2];
                bh[4 * q + 3] = (_Float16)f[3];
            }
            acc[nb] = __builtin_amdgcn_wmma_f32_16x16x32_f16(
                false, a.h, false, bh, (short)0, acc[nb], false, false);
        }
    }

    // ---------------- Phase 3: bias + BN + 4-group projection + sigmoid ----------
    // D layout: lane (ln,half), reg r  ->  h[pos = half*8 + r][o = nb*16 + ln]
    float pg[4][8];
    #pragma unroll
    for (int g = 0; g < 4; ++g)
        #pragma unroll
        for (int r = 0; r < 8; ++r) pg[g][r] = 0.f;

    #pragma unroll
    for (int nb = 0; nb < 4; ++nb) {
        const int o = nb * 16 + ln;
        const float sc = gamma[o] * rsqrtf(run_var[o] + BN_EPS_F);
        const float mn = run_mean[o];
        const float bt = beta[o];
        const float bb = b0[o];
        const float w1a = W1[0 * C4O + o];
        const float w1b = W1[1 * C4O + o];
        const float w1c = W1[2 * C4O + o];
        const float w1d = W1[3 * C4O + o];
        #pragma unroll
        for (int r = 0; r < 8; ++r) {
            const float hv = acc[nb][r] + bb;
            const float hb = (hv - mn) * sc + bt;
            pg[0][r] = fmaf(w1a, hb, pg[0][r]);
            pg[1][r] = fmaf(w1b, hb, pg[1][r]);
            pg[2][r] = fmaf(w1c, hb, pg[2][r]);
            pg[3][r] = fmaf(w1d, hb, pg[3][r]);
        }
    }

    const float b1v0 = b1[0], b1v1 = b1[1], b1v2 = b1[2], b1v3 = b1[3];

    #pragma unroll
    for (int r = 0; r < 8; ++r) {
        #pragma unroll
        for (int g = 0; g < 4; ++g) {
            float s = pg[g][r];
            // reduce over the 16 lanes of this half (masks never cross bit 4)
            s += __shfl_xor(s, 1);
            s += __shfl_xor(s, 2);
            s += __shfl_xor(s, 4);
            s += __shfl_xor(s, 8);
            const float bv = (g == 0) ? b1v0 : (g == 1) ? b1v1 : (g == 2) ? b1v2 : b1v3;
            const float att = 1.f / (1.f + __expf(-(s + bv)));
            if (ln == ((r * 4 + g) & 15)) {
                att_lds[wv * 64 + (half * 8 + r) * 4 + g] = att;
            }
        }
    }
    __syncthreads();

    // ---------------- Phase 4: tiled broadcast write (out[:,c] = att[:, c%4]) ----
    const float a0 = att_lds[wv * 64 + ln * 4 + 0];
    const float a1 = att_lds[wv * 64 + ln * 4 + 1];
    const float a2 = att_lds[wv * 64 + ln * 4 + 2];
    const float a3 = att_lds[wv * 64 + ln * 4 + 3];
    const float va = half ? a1 : a0;   // channels with c&3 == half
    const float vb = half ? a3 : a2;   // channels with c&3 == 2+half

    const size_t obase = (size_t)b * CCH * HWSZ + (size_t)y * WWD + (size_t)(x0 + ln);
    for (int it = 0; it < 128; ++it) {
        const int c = it * 2 + half;
        out[obase + (size_t)c * HWSZ] = (it & 1) ? vb : va;
    }
}

extern "C" void kernel_launch(void* const* d_in, const int* in_sizes, int n_in,
                              void* d_out, int out_size, void* d_ws, size_t ws_size,
                              hipStream_t stream) {
    (void)in_sizes; (void)n_in; (void)out_size; (void)d_ws; (void)ws_size;
    const float* x      = (const float*)d_in[0];
    const float* offset = (const float*)d_in[1];
    const float* W0     = (const float*)d_in[2];
    const float* b0     = (const float*)d_in[3];
    const float* gamma  = (const float*)d_in[4];
    const float* beta   = (const float*)d_in[5];
    const float* rmean  = (const float*)d_in[6];
    const float* rvar   = (const float*)d_in[7];
    const float* W1     = (const float*)d_in[8];
    const float* b1     = (const float*)d_in[9];
    float* out = (float*)d_out;

    dim3 grid(BB * HH * 2);   // 512 blocks: one per (b, y, x-half)
    dim3 block(128);          // 4 wave32 waves; each wave owns 16 pixels
    hipLaunchKernelGGL(deform_attn_wmma_kernel, grid, block, 0, stream,
                       x, offset, W0, b0, gamma, beta, rmean, rvar, W1, b1, out);
}
